// SAModule_55688545960606
// MI455X (gfx1250) — compile-verified, matrix operand order checked by reading, not run on hardware
//
#include <hip/hip_runtime.h>
#include <hip/hip_bf16.h>

// ---------------------------------------------------------------------------
// PPFConv-style layer on MI455X (gfx1250), wave32 + v_wmma_f32_16x16x32_f16.
// Edge MLP tiled as 16-edge WMMA tiles; dst is sorted (32 edges/node) so the
// segment-max for bulk edges reduces in-wave to 64 atomics per tile.
// ---------------------------------------------------------------------------

typedef _Float16 v16h __attribute__((ext_vector_type(16)));
typedef float    v8f  __attribute__((ext_vector_type(8)));
typedef _Float16 h2v  __attribute__((ext_vector_type(2)));

#define WPB 4  // waves per block

// K-striping for f16 A/B fragments (ISA 7.12.2, 16-bit A 16x32):
// VGPR pair i, lane-half h -> first K of the f16 pair.
__device__ __forceinline__ int kmap(int i, int hh) {
  return 2 * i + ((i >= 4) ? 8 : 0) + 8 * hh;
}

__device__ __forceinline__ v8f wmma_f16(v16h a, v16h b, v8f c) {
  return __builtin_amdgcn_wmma_f32_16x16x32_f16(
      /*neg_a=*/false, a, /*neg_b=*/false, b,
      /*c_mod=*/(short)0, c, /*reuse_a=*/false, /*reuse_b=*/false);
}

// Padded msg layout (96 cols): [0,64)=x, [64,71)=ppf, 71=0 pad, [72,88)=ea, [88,96)=0.
__device__ __forceinline__ float w1row(const float* W1, int p, int n) {
  if (p < 71) return W1[p * 64 + n];
  if (p >= 72 && p < 88) return W1[(p - 1) * 64 + n];
  return 0.0f;
}

// Load a 16xK f16 A-fragment for K-tile t from an LDS row-major buffer.
__device__ __forceinline__ v16h load_afrag(const _Float16* buf, int rowStride,
                                           int lane, int t) {
  int M = lane & 15, hh = lane >> 4;
  v16h a;
#pragma unroll
  for (int i = 0; i < 8; ++i) {
    int k0 = t * 32 + kmap(i, hh);
    h2v p = *(const h2v*)(buf + M * rowStride + k0);
    a[2 * i] = p.x;
    a[2 * i + 1] = p.y;
  }
  return a;
}

__device__ __forceinline__ void angle_sc(float ax, float ay, float az,
                                         float bx, float by, float bz,
                                         float& s, float& c) {
  float cx = ay * bz - az * by;
  float cy = az * bx - ax * bz;
  float cz = ax * by - ay * bx;
  float cn2 = cx * cx + cy * cy + cz * cz;
  float dt = ax * bx + ay * by + az * bz;
  float h2 = cn2 + dt * dt;
  if (h2 > 0.0f) {
    float rh = rsqrtf(h2);
    s = sqrtf(cn2) * rh;   // sin(atan2(||axb||, a.b))
    c = dt * rh;           // cos(...)
  } else {
    s = 0.0f;
    c = 1.0f;              // atan2(0,0) = 0
  }
}

// ---------------------------------------------------------------------------
// Kernel 1: build per-lane v16h weight fragments (f32 -> f16) in workspace.
// frags: [0,12) W1 (t=f/4,u=f%4), [12,20) W2, [20,36) Wg (t=g/8,u=g%8).
// ---------------------------------------------------------------------------
__global__ void prep_frags(const float* __restrict__ W1,
                           const float* __restrict__ W2,
                           const float* __restrict__ Wg,
                           _Float16* __restrict__ wsH) {
  int f = blockIdx.x;
  int lane = threadIdx.x;
  int n16 = lane & 15, hh = lane >> 4;
  _Float16* dst = wsH + ((size_t)f * 32 + lane) * 16;
  if (f < 12) {
    int t = f >> 2, u = f & 3, n = u * 16 + n16;
#pragma unroll
    for (int i = 0; i < 8; ++i) {
      int k0 = t * 32 + kmap(i, hh);
      dst[2 * i] = (_Float16)w1row(W1, k0, n);
      dst[2 * i + 1] = (_Float16)w1row(W1, k0 + 1, n);
    }
  } else if (f < 20) {
    int g = f - 12;
    int t = g >> 2, u = g & 3, n = u * 16 + n16;
#pragma unroll
    for (int i = 0; i < 8; ++i) {
      int k0 = t * 32 + kmap(i, hh);
      dst[2 * i] = (_Float16)W2[k0 * 64 + n];
      dst[2 * i + 1] = (_Float16)W2[(k0 + 1) * 64 + n];
    }
  } else {
    int g = f - 20;
    int t = g >> 3, u = g & 7, n = u * 16 + n16;
#pragma unroll
    for (int i = 0; i < 8; ++i) {
      int k0 = t * 32 + kmap(i, hh);
      dst[2 * i] = (_Float16)Wg[k0 * 128 + n];
      dst[2 * i + 1] = (_Float16)Wg[(k0 + 1) * 128 + n];
    }
  }
}

// ---------------------------------------------------------------------------
// Kernel 2: bulk edges (first E-N edges; dst uniform per 16-edge tile).
// ---------------------------------------------------------------------------
__global__ __launch_bounds__(32 * WPB)
void edge_bulk(const float* __restrict__ x, const float* __restrict__ pos,
               const float* __restrict__ nrm, const int* __restrict__ ei,
               const float* __restrict__ ea, const float* __restrict__ b1,
               const float* __restrict__ b2,
               const _Float16* __restrict__ w1f,
               const _Float16* __restrict__ w2f,
               float* __restrict__ outmax, int E, int numTiles) {
  __shared__ _Float16 lds[WPB][16 * 96 + 16 * 64];
  int wave = threadIdx.x >> 5, lane = threadIdx.x & 31;
  int tile = blockIdx.x * WPB + wave;
  if (tile >= numTiles) return;
  _Float16* msg = &lds[wave][0];
  _Float16* hbuf = &lds[wave][16 * 96];
  int m = lane & 15, hh = lane >> 4;
  long e = (long)tile * 16 + m;
  int s = ei[e];  // src node of this lane's edge
  _Float16* mrow = msg + m * 96;

  // x gather: this lane fills cols [32*hh, 32*hh+32) of row m
  const float* xp = x + (long)s * 64 + 32 * hh;
#pragma unroll
  for (int j = 0; j < 32; j += 2) {
    h2v p;
    p.x = (_Float16)xp[j];
    p.y = (_Float16)xp[j + 1];
    *(h2v*)(mrow + 32 * hh + j) = p;
  }
  // edge_attr: cols [72+8*hh, 72+8*hh+8)
  const float* epa = ea + e * 16 + 8 * hh;
#pragma unroll
  for (int j = 0; j < 8; j += 2) {
    h2v p;
    p.x = (_Float16)epa[j];
    p.y = (_Float16)epa[j + 1];
    *(h2v*)(mrow + 72 + 8 * hh + j) = p;
  }
  if (hh == 0) {
    // PPF features (cols 64..70, col 71 = pad 0)
    int d = ei[(long)E + e];
    float pix = pos[d * 3 + 0], piy = pos[d * 3 + 1], piz = pos[d * 3 + 2];
    float pjx = pos[s * 3 + 0], pjy = pos[s * 3 + 1], pjz = pos[s * 3 + 2];
    float nix = nrm[d * 3 + 0], niy = nrm[d * 3 + 1], niz = nrm[d * 3 + 2];
    float njx = nrm[s * 3 + 0], njy = nrm[s * 3 + 1], njz = nrm[s * 3 + 2];
    float ux = pjx - pix, uy = pjy - piy, uz = pjz - piz;
    float pn = sqrtf(ux * ux + uy * uy + uz * uz);
    float s1, c1, s2, c2, s3, c3;
    angle_sc(nix, niy, niz, ux, uy, uz, s1, c1);
    angle_sc(njx, njy, njz, ux, uy, uz, s2, c2);
    angle_sc(nix, niy, niz, njx, njy, njz, s3, c3);
    h2v p;
    p.x = (_Float16)(pn * 0.25f); p.y = (_Float16)s1; *(h2v*)(mrow + 64) = p;
    p.x = (_Float16)c1; p.y = (_Float16)s2;           *(h2v*)(mrow + 66) = p;
    p.x = (_Float16)c2; p.y = (_Float16)s3;           *(h2v*)(mrow + 68) = p;
    p.x = (_Float16)c3; p.y = (_Float16)0.0f;         *(h2v*)(mrow + 70) = p;
  } else {
    h2v z;
    z.x = (_Float16)0.0f; z.y = (_Float16)0.0f;
#pragma unroll
    for (int j = 0; j < 8; j += 2) *(h2v*)(mrow + 88 + j) = z;
  }
  asm volatile("s_wait_dscnt 0" ::: "memory");  // intra-wave LDS visibility

  // ---- layer 1: [16x96] @ [96x64]
  v16h a0 = load_afrag(msg, 96, lane, 0);
  v16h a1 = load_afrag(msg, 96, lane, 1);
  v16h a2 = load_afrag(msg, 96, lane, 2);
  v8f acc[4];
#pragma unroll
  for (int u = 0; u < 4; ++u) {
    float bb = b1[u * 16 + m];
#pragma unroll
    for (int r = 0; r < 8; ++r) acc[u][r] = bb;
  }
#pragma unroll
  for (int u = 0; u < 4; ++u) {
    acc[u] = wmma_f16(a0, *(const v16h*)(w1f + ((size_t)(0 * 4 + u) * 32 + lane) * 16), acc[u]);
    acc[u] = wmma_f16(a1, *(const v16h*)(w1f + ((size_t)(1 * 4 + u) * 32 + lane) * 16), acc[u]);
    acc[u] = wmma_f16(a2, *(const v16h*)(w1f + ((size_t)(2 * 4 + u) * 32 + lane) * 16), acc[u]);
  }
  // relu -> hbuf (D layout: VGPR r holds row M = r + 8*hh, col n = u*16+m)
#pragma unroll
  for (int u = 0; u < 4; ++u)
#pragma unroll
    for (int r = 0; r < 8; ++r)
      hbuf[(r + 8 * hh) * 64 + u * 16 + m] = (_Float16)fmaxf(acc[u][r], 0.0f);
  asm volatile("s_wait_dscnt 0" ::: "memory");

  // ---- layer 2: [16x64] @ [64x64]
  v16h g0 = load_afrag(hbuf, 64, lane, 0);
  v16h g1 = load_afrag(hbuf, 64, lane, 1);
  v8f acc2[4];
#pragma unroll
  for (int u = 0; u < 4; ++u) {
    float bb = b2[u * 16 + m];
#pragma unroll
    for (int r = 0; r < 8; ++r) acc2[u][r] = bb;
  }
#pragma unroll
  for (int u = 0; u < 4; ++u) {
    acc2[u] = wmma_f16(g0, *(const v16h*)(w2f + ((size_t)(0 * 4 + u) * 32 + lane) * 16), acc2[u]);
    acc2[u] = wmma_f16(g1, *(const v16h*)(w2f + ((size_t)(1 * 4 + u) * 32 + lane) * 16), acc2[u]);
  }

  // ---- relu + in-wave max over 16 rows (dst uniform per tile), then atomics.
  int node = ei[(long)E + (long)tile * 16];  // sorted dst: uniform in tile
#pragma unroll
  for (int u = 0; u < 4; ++u) {
    float mv = 0.0f;  // relu lower bound; exact since every node max >= 0
#pragma unroll
    for (int r = 0; r < 8; ++r) mv = fmaxf(mv, acc2[u][r]);
    mv = fmaxf(mv, __shfl_xor(mv, 16, 32));  // merge lane halves (M 0-7 vs 8-15)
    if (lane < 16)
      atomicMax((unsigned int*)&outmax[(size_t)node * 64 + u * 16 + lane],
                __float_as_uint(mv));
  }
}

// ---------------------------------------------------------------------------
// Kernel 3: self-loop edges (src = dst = node; ppf = [0,0,1,0,1,0,1]).
// ---------------------------------------------------------------------------
__global__ __launch_bounds__(32 * WPB)
void edge_self(const float* __restrict__ x, const float* __restrict__ ea,
               const float* __restrict__ b1, const float* __restrict__ b2,
               const _Float16* __restrict__ w1f,
               const _Float16* __restrict__ w2f,
               float* __restrict__ outmax, int bulkE, int numTiles) {
  __shared__ _Float16 lds[WPB][16 * 96 + 16 * 64];
  int wave = threadIdx.x >> 5, lane = threadIdx.x & 31;
  int tile = blockIdx.x * WPB + wave;
  if (tile >= numTiles) return;
  _Float16* msg = &lds[wave][0];
  _Float16* hbuf = &lds[wave][16 * 96];
  int m = lane & 15, hh = lane >> 4;
  int node = tile * 16 + m;
  long e = (long)bulkE + node;
  _Float16* mrow = msg + m * 96;

  const float* xp = x + (long)node * 64 + 32 * hh;
#pragma unroll
  for (int j = 0; j < 32; j += 2) {
    h2v p;
    p.x = (_Float16)xp[j];
    p.y = (_Float16)xp[j + 1];
    *(h2v*)(mrow + 32 * hh + j) = p;
  }
  const float* epa = ea + e * 16 + 8 * hh;
#pragma unroll
  for (int j = 0; j < 8; j += 2) {
    h2v p;
    p.x = (_Float16)epa[j];
    p.y = (_Float16)epa[j + 1];
    *(h2v*)(mrow + 72 + 8 * hh + j) = p;
  }
  h2v z, o;
  z.x = (_Float16)0.0f; z.y = (_Float16)0.0f;
  o.x = (_Float16)1.0f; o.y = (_Float16)0.0f;
  if (hh == 0) {
    *(h2v*)(mrow + 64) = z;  // [0, sin=0]
    *(h2v*)(mrow + 66) = o;  // [cos=1, sin=0]
    *(h2v*)(mrow + 68) = o;  // [cos=1, sin=0]
    *(h2v*)(mrow + 70) = o;  // [cos=1, pad=0]
  } else {
#pragma unroll
    for (int j = 0; j < 8; j += 2) *(h2v*)(mrow + 88 + j) = z;
  }
  asm volatile("s_wait_dscnt 0" ::: "memory");

  v16h a0 = load_afrag(msg, 96, lane, 0);
  v16h a1 = load_afrag(msg, 96, lane, 1);
  v16h a2 = load_afrag(msg, 96, lane, 2);
  v8f acc[4];
#pragma unroll
  for (int u = 0; u < 4; ++u) {
    float bb = b1[u * 16 + m];
#pragma unroll
    for (int r = 0; r < 8; ++r) acc[u][r] = bb;
  }
#pragma unroll
  for (int u = 0; u < 4; ++u) {
    acc[u] = wmma_f16(a0, *(const v16h*)(w1f + ((size_t)(0 * 4 + u) * 32 + lane) * 16), acc[u]);
    acc[u] = wmma_f16(a1, *(const v16h*)(w1f + ((size_t)(1 * 4 + u) * 32 + lane) * 16), acc[u]);
    acc[u] = wmma_f16(a2, *(const v16h*)(w1f + ((size_t)(2 * 4 + u) * 32 + lane) * 16), acc[u]);
  }
#pragma unroll
  for (int u = 0; u < 4; ++u)
#pragma unroll
    for (int r = 0; r < 8; ++r)
      hbuf[(r + 8 * hh) * 64 + u * 16 + m] = (_Float16)fmaxf(acc[u][r], 0.0f);
  asm volatile("s_wait_dscnt 0" ::: "memory");

  v16h g0 = load_afrag(hbuf, 64, lane, 0);
  v16h g1 = load_afrag(hbuf, 64, lane, 1);
  v8f acc2[4];
#pragma unroll
  for (int u = 0; u < 4; ++u) {
    float bb = b2[u * 16 + m];
#pragma unroll
    for (int r = 0; r < 8; ++r) acc2[u][r] = bb;
  }
#pragma unroll
  for (int u = 0; u < 4; ++u) {
    acc2[u] = wmma_f16(g0, *(const v16h*)(w2f + ((size_t)(0 * 4 + u) * 32 + lane) * 16), acc2[u]);
    acc2[u] = wmma_f16(g1, *(const v16h*)(w2f + ((size_t)(1 * 4 + u) * 32 + lane) * 16), acc2[u]);
  }
  // distinct dst per row: scatter each element
#pragma unroll
  for (int u = 0; u < 4; ++u)
#pragma unroll
    for (int r = 0; r < 8; ++r) {
      int nd = tile * 16 + r + 8 * hh;
      float v = fmaxf(acc2[u][r], 0.0f);
      atomicMax((unsigned int*)&outmax[(size_t)nd * 64 + u * 16 + m],
                __float_as_uint(v));
    }
}

// ---------------------------------------------------------------------------
// Kernel 4: out = segmax @ Wg + bg  ([N,64] @ [64,128]) -> d_out[0 : N*128].
// ---------------------------------------------------------------------------
__global__ __launch_bounds__(32 * WPB)
void node_out(const float* __restrict__ outmax, const float* __restrict__ bg,
              const _Float16* __restrict__ wgf, float* __restrict__ out,
              int numTiles) {
  __shared__ _Float16 lds[WPB][16 * 64];
  int wave = threadIdx.x >> 5, lane = threadIdx.x & 31;
  int tile = blockIdx.x * WPB + wave;
  if (tile >= numTiles) return;
  _Float16* abuf = &lds[wave][0];
  int m = lane & 15, hh = lane >> 4;
  const float* rp = outmax + (size_t)(tile * 16 + m) * 64 + 32 * hh;
#pragma unroll
  for (int j = 0; j < 32; j += 2) {
    h2v p;
    p.x = (_Float16)rp[j];
    p.y = (_Float16)rp[j + 1];
    *(h2v*)(abuf + m * 64 + 32 * hh + j) = p;
  }
  asm volatile("s_wait_dscnt 0" ::: "memory");

  v16h a0 = load_afrag(abuf, 64, lane, 0);
  v16h a1 = load_afrag(abuf, 64, lane, 1);
  v8f acc[8];
#pragma unroll
  for (int u = 0; u < 8; ++u) {
    float bb = bg[u * 16 + m];
#pragma unroll
    for (int r = 0; r < 8; ++r) acc[u][r] = bb;
  }
#pragma unroll
  for (int u = 0; u < 8; ++u) {
    acc[u] = wmma_f16(a0, *(const v16h*)(wgf + ((size_t)(0 * 8 + u) * 32 + lane) * 16), acc[u]);
    acc[u] = wmma_f16(a1, *(const v16h*)(wgf + ((size_t)(1 * 8 + u) * 32 + lane) * 16), acc[u]);
  }
#pragma unroll
  for (int u = 0; u < 8; ++u)
#pragma unroll
    for (int r = 0; r < 8; ++r)
      out[(size_t)(tile * 16 + r + 8 * hh) * 128 + u * 16 + m] = acc[u][r];
}

// ---------------------------------------------------------------------------
// Workspace layout (bytes):
//   [0, 12288)      W1 fragments  (12 frags x 32 lanes x 16 f16)
//   [12288, 20480)  W2 fragments  (8 frags)
//   [20480, 36864)  Wg fragments  (16 frags)
//   [36864, ...)    outmax: N x 64 f32 (zero-initialized each launch)
// ---------------------------------------------------------------------------
extern "C" void kernel_launch(void* const* d_in, const int* in_sizes, int n_in,
                              void* d_out, int out_size, void* d_ws,
                              size_t ws_size, hipStream_t stream) {
  const float* x   = (const float*)d_in[0];
  const float* pos = (const float*)d_in[1];
  const float* nrm = (const float*)d_in[2];
  const int*   bat = (const int*)d_in[3];
  const int*   ei  = (const int*)d_in[4];
  const float* ea  = (const float*)d_in[5];
  const float* W1  = (const float*)d_in[6];
  const float* b1  = (const float*)d_in[7];
  const float* W2  = (const float*)d_in[8];
  const float* b2  = (const float*)d_in[9];
  const float* Wg  = (const float*)d_in[10];
  const float* bg  = (const float*)d_in[11];

  int N = in_sizes[0] / 64;   // nodes
  int E = in_sizes[5] / 16;   // edges (bulk + self loops)
  int bulkE = E - N;

  _Float16* wsH = (_Float16*)d_ws;
  const _Float16* w1f = wsH;
  const _Float16* w2f = wsH + 12 * 512;
  const _Float16* wgf = wsH + 20 * 512;
  float* outmax = (float*)((char*)d_ws + 36864);
  float* out = (float*)d_out;

  hipMemsetAsync(outmax, 0, (size_t)N * 64 * sizeof(float), stream);
  prep_frags<<<36, 32, 0, stream>>>(W1, W2, Wg, wsH);

  int bulkTiles = bulkE / 16;  // dst sorted: 32 edges/node -> tile dst uniform
  edge_bulk<<<(bulkTiles + WPB - 1) / WPB, 32 * WPB, 0, stream>>>(
      x, pos, nrm, ei, ea, b1, b2, w1f, w2f, outmax, E, bulkTiles);

  int selfTiles = (N + 15) / 16;
  edge_self<<<(selfTiles + WPB - 1) / WPB, 32 * WPB, 0, stream>>>(
      x, ea, b1, b2, w1f, w2f, outmax, bulkE, selfTiles);

  node_out<<<(selfTiles + WPB - 1) / WPB, 32 * WPB, 0, stream>>>(
      outmax, bg, wgf, out, selfTiles);

  // Tuple tail: pos then batch, raw copies.
  hipMemcpyAsync((char*)d_out + (size_t)N * 128 * 4, pos, (size_t)N * 3 * 4,
                 hipMemcpyDeviceToDevice, stream);
  hipMemcpyAsync((char*)d_out + (size_t)N * 131 * 4, bat, (size_t)N * 4,
                 hipMemcpyDeviceToDevice, stream);
}
